// SSIMLoss_30459908063528
// MI455X (gfx1250) — matrix operand order repaired and checked by reading, not compiled, not run on hardware
//
#include <hip/hip_runtime.h>

// SSIM loss, fused single-pass:
//   stage 1: load pred/target 74x74 halo region -> 5 derived f16 images in LDS
//   stage 2: horizontal 11-tap blur as V_WMMA_F32_16X16X32_F16 (A=data, B=banded gaussian)
//            result stored transposed (f16) in LDS
//   stage 3: vertical blur as WMMA (A=banded gaussian, B=transposed data),
//            elementwise SSIM map in fp32, per-lane accumulate
//   reduce : LDS tree -> per-workgroup partial -> finish kernel (fixed order)

typedef _Float16 f16;
typedef __attribute__((ext_vector_type(16))) _Float16 v16h;
typedef __attribute__((ext_vector_type(8)))  _Float16 v8h;
typedef __attribute__((ext_vector_type(8)))  float    v8f;

#define IMG_H   512
#define IMG_W   512
#define NBATCH  64
#define RP      80                       // padded region pitch (halves) and padded rows
#define REG_HALVES (RP * RP)             // 6400 per derived image
#define HST_HALVES (64 * RP)             // 5120 per image (64 cols x 80 rows, transposed)
#define SMEM_BYTES (5 * REG_HALVES * 2 + 5 * HST_HALVES * 2)   // 64000 + 51200 = 115200
#define GPAD 16                          // zero padding on each side of the 11-tap kernel

__global__ void __launch_bounds__(256)
ssim_main_kernel(const float* __restrict__ pred,
                 const float* __restrict__ targ,
                 const float* __restrict__ window,   // [1,1,11,11]
                 float* __restrict__ partial)
{
    extern __shared__ __align__(128) char smem[];
    f16* reg0 = (f16*)smem;                              // 5 x 6400 halves
    f16* hst0 = (f16*)(smem + 5 * REG_HALVES * 2);       // 5 x 5120 halves
    __shared__ float gPad[48];   // zeros except gPad[16..26] = 1-D gaussian

    const int tid  = threadIdx.x;
    const int lane = tid & 31;
    const int wave = tid >> 5;
    const bool low = (lane < 16);
    const int  mN  = lane & 15;

    const int C0 = blockIdx.x * 64;
    const int R0 = blockIdx.y * 64;
    const int img = blockIdx.z;
    const float* pbase = pred + (size_t)img * IMG_H * IMG_W;
    const float* tbase = targ + (size_t)img * IMG_H * IMG_W;

    // Recover separable 1-D kernel: window = outer(g,g), sum(g)=1 -> row sums = g.
    // Padded table so fragment builds need no range checks (index always in [1,47]).
    if (tid < 48) {
        float v = 0.f;
        if (tid >= GPAD && tid < GPAD + 11) {
            const float* wrow = window + (tid - GPAD) * 11;
            float s = 0.f;
            #pragma unroll
            for (int j = 0; j < 11; ++j) s += wrow[j];
            v = s;
        }
        gPad[tid] = v;
    }

    // ---- Stage 1: fill 5 derived images (f16) over 80x80 padded region (74x74 valid)
    for (int idx = tid; idx < RP * RP; idx += 256) {
        const int rr = idx / RP, cc = idx % RP;
        const int r = R0 - 5 + rr, c = C0 - 5 + cc;
        float p = 0.f, t = 0.f;
        if (rr < 74 && cc < 74 && r >= 0 && r < IMG_H && c >= 0 && c < IMG_W) {
            p = pbase[r * IMG_W + c];
            t = tbase[r * IMG_W + c];
        }
        reg0[0 * REG_HALVES + idx] = (f16)p;
        reg0[1 * REG_HALVES + idx] = (f16)t;
        reg0[2 * REG_HALVES + idx] = (f16)(p * p);
        reg0[3 * REG_HALVES + idx] = (f16)(t * t);
        reg0[4 * REG_HALVES + idx] = (f16)(p * t);
    }
    __syncthreads();

    // Horizontal banded-weight B fragment: B[k][n] = g[k-n] for 0<=k-n<=10, else 0.
// B layout (32x16 f16): lane n<16 -> N=n, K={0..7,16..23}; lane>=16 -> N=n-16, K={8..15,24..31}.
    v16h bH;
    {
        const int kb = (low ? 0 : 8) - mN + GPAD;
        f16 tmp[16];
        #pragma unroll
        for (int i = 0; i < 16; ++i) tmp[i] = (f16)gPad[kb + (i < 8 ? i : i + 8)];
        #pragma unroll
        for (int i = 0; i < 16; ++i) bH[i] = tmp[i];
    }

    // ---- Stage 2: horizontal blur, 100 WMMAs (5 images x 5 row-groups x 4 col-groups)
    for (int task = wave; task < 100; task += 8) {
        const int s   = task / 20;
        const int rem = task % 20;
        const int rg  = rem >> 2;       // 0..4  (rows 16*rg .. 16*rg+15)
        const int j   = rem & 3;        // 0..3  (out cols 16*j .. 16*j+15)
        const f16* regS = reg0 + s * REG_HALVES;

        // A layout (16x32 f16): lane<16 -> M=lane, K={0..7,16..23}; lane>=16 -> K={8..15,24..31}
        const int arow = rg * 16 + mN;
        const int c0   = low ? 0 : 8;
        const int base = arow * RP + j * 16;
        v8h loH = *(const v8h*)(regS + base + c0);
        v8h hiH = *(const v8h*)(regS + base + c0 + 16);
        v16h a = __builtin_shufflevector(loH, hiH, 0,1,2,3,4,5,6,7,8,9,10,11,12,13,14,15);

        v8f c = {0.f,0.f,0.f,0.f,0.f,0.f,0.f,0.f};
        c = __builtin_amdgcn_wmma_f32_16x16x32_f16(false, a, false, bH, (short)0, c, false, false);

        // Store transposed: hst[s][col][row], f16.  D elem e: M = e + (low?0:8), N = mN.
        f16* hs = hst0 + s * HST_HALVES;
        const int colo  = j * 16 + mN;
        const int rbase = rg * 16 + (low ? 0 : 8);
        #pragma unroll
        for (int e = 0; e < 8; ++e)
            hs[colo * RP + rbase + e] = (f16)c[e];
    }
    __syncthreads();

    // Vertical banded-weight A fragment: A[m][k] = g[k-m] for 0<=k-m<=10, else 0.
    v16h aV;
    {
        f16 tmp[16];
        #pragma unroll
        for (int i = 0; i < 16; ++i) {
            const int K = low ? i : i + 16;
            tmp[i] = (f16)gPad[K - mN + GPAD];
        }
        #pragma unroll
        for (int i = 0; i < 16; ++i) aV[i] = tmp[i];
    }

    // ---- Stage 3: vertical blur (80 WMMAs) + SSIM map + per-lane accumulate
    float acc = 0.f;
    const float Cc1 = 0.0001f, Cc2 = 0.0009f;   // 0.01^2, 0.03^2

    #pragma unroll
    for (int tt = 0; tt < 2; ++tt) {
        const int t  = wave + tt * 8;   // 0..15 output 16x16 tiles
        const int tr = t >> 2, tc = t & 3;
        const int colo = tc * 16 + mN;
        const int rofs = tr * 16 + (low ? 0 : 16);
        v8f d[5];
        #pragma unroll
        for (int s = 0; s < 5; ++s) {
            const f16* hs = hst0 + s * HST_HALVES;
            v16h bfrag = *(const v16h*)(hs + colo * RP + rofs);
            v8f cz = {0.f,0.f,0.f,0.f,0.f,0.f,0.f,0.f};
            d[s] = __builtin_amdgcn_wmma_f32_16x16x32_f16(false, aV, false, bfrag,
                                                          (short)0, cz, false, false);
        }
        #pragma unroll
        for (int e = 0; e < 8; ++e) {
            const float mu1 = d[0][e], mu2 = d[1][e];
            const float m11 = mu1 * mu1, m22 = mu2 * mu2, m12 = mu1 * mu2;
            const float s1  = d[2][e] - m11;
            const float s2  = d[3][e] - m22;
            const float s12 = d[4][e] - m12;
            const float num = (2.f * m12 + Cc1) * (2.f * s12 + Cc2);
            const float den = (m11 + m22 + Cc1) * (s1 + s2 + Cc2);
            acc += num / den;
        }
    }

    // ---- Fixed-order workgroup reduction (reuse region LDS; stage 3 only reads hst)
    float* red = (float*)smem;
    red[tid] = acc;
    __syncthreads();
    #pragma unroll
    for (int k = 128; k > 0; k >>= 1) {
        if (tid < k) red[tid] += red[tid + k];
        __syncthreads();
    }
    if (tid == 0)
        partial[blockIdx.z * 64 + blockIdx.y * 8 + blockIdx.x] = red[0];
}

__global__ void __launch_bounds__(256)
ssim_finish_kernel(const float* __restrict__ partial, float* __restrict__ out)
{
    __shared__ float red[256];
    const int tid = threadIdx.x;
    float s = 0.f;
    for (int i = tid; i < 4096; i += 256) s += partial[i];   // fixed order
    red[tid] = s;
    __syncthreads();
    #pragma unroll
    for (int k = 128; k > 0; k >>= 1) {
        if (tid < k) red[tid] += red[tid + k];
        __syncthreads();
    }
    if (tid == 0)
        out[0] = 1.0f - red[0] * (1.0f / 16777216.0f);       // N = 64*512*512
}

extern "C" void kernel_launch(void* const* d_in, const int* in_sizes, int n_in,
                              void* d_out, int out_size, void* d_ws, size_t ws_size,
                              hipStream_t stream)
{
    const float* pred   = (const float*)d_in[0];
    const float* targ   = (const float*)d_in[1];
    const float* window = (const float*)d_in[2];   // [1,1,11,11] fp32
    float* partial = (float*)d_ws;                 // 4096 floats; every slot written each launch
    float* out = (float*)d_out;

    // Allow >64KB dynamic LDS (WGP has 320KB). Not a stream op; capture-safe.
    (void)hipFuncSetAttribute((const void*)ssim_main_kernel,
                              hipFuncAttributeMaxDynamicSharedMemorySize, SMEM_BYTES);

    dim3 grid(IMG_W / 64, IMG_H / 64, NBATCH);     // 8 x 8 x 64 = 4096 workgroups
    ssim_main_kernel<<<grid, dim3(256), SMEM_BYTES, stream>>>(pred, targ, window, partial);
    ssim_finish_kernel<<<dim3(1), dim3(256), 0, stream>>>(partial, out);
}
